// VectorQuantizerEMA_21311627723518
// MI455X (gfx1250) — matrix operand order repaired and checked by reading, not compiled, not run on hardware
//
#include <hip/hip_runtime.h>
#include <hip/hip_bf16.h>
#include <math.h>

// Problem constants (from reference): B,H,W,D,K = 32,16,16,256,8192
#define DIM   256
#define KC    8192
#define NTOK  8192          // 32*16*16 flattened tokens
#define DECAY_F 0.99f
#define ONE_MINUS_DECAY (1.0f - 0.99f)

typedef __attribute__((ext_vector_type(16))) __bf16 v16bf;
typedef __attribute__((ext_vector_type(8)))  float  v8f;

// ---------------------------------------------------------------------------
// wsq[k] = sum_d w[d][k]^2   (coalesced across threads; w is [D][K] row-major)
// ---------------------------------------------------------------------------
__global__ __launch_bounds__(256) void vq_wsq(const float* __restrict__ w,
                                              float* __restrict__ wsq) {
  int k = blockIdx.x * 256 + threadIdx.x;
  float s = 0.f;
  for (int d = 0; d < DIM; ++d) {
    float v = w[(size_t)d * KC + k];
    s += v * v;
  }
  wsq[k] = s;
}

// ---------------------------------------------------------------------------
// One-time: transpose w -> [K][D] and split fp32 into hi/lo bf16 arrays.
// wtHi[k*256+d] = bf16(w[d][k]); wtLo = bf16(w[d][k] - fp32(hi)).
// ---------------------------------------------------------------------------
__global__ __launch_bounds__(256) void vq_cvt_w(const float* __restrict__ w,
                                                __bf16* __restrict__ wtHi) {
  int i = blockIdx.x * 256 + threadIdx.x;   // i in [0, K*D)
  int k = i >> 8;                           // D == 256
  int d = i & 255;
  float f = w[(size_t)d * KC + k];
  __bf16 h = (__bf16)f;
  wtHi[i] = h;
  wtHi[(size_t)KC * DIM + i] = (__bf16)(f - (float)h);   // lo array
}

// ---------------------------------------------------------------------------
// new_ema_w := DECAY * ema_w   (scatter target for the dw accumulation)
// ---------------------------------------------------------------------------
__global__ __launch_bounds__(256) void vq_init_emaw(const float* __restrict__ emaW,
                                                    float* __restrict__ outEmaW) {
  int i = blockIdx.x * 256 + threadIdx.x;
  outEmaW[i] = DECAY_F * emaW[i];
}

// ---------------------------------------------------------------------------
// Fused distance-GEMM + argmin.  Block = 64 tokens (4 bands of 16 rows).
// argmin target: dist = wsq[k] - 2 * (x . w[:,k])   (||x||^2 row-constant).
// bf16 hi/lo split: x.w ~= xh.wh + xh.wl + xl.wh  (fp32 accumulate).
//
// A fragments (16-bit A 16x32, wave32) staged once per block into LDS:
//   lane L: row M = L&15, element e -> K = (L>=16?16:0) + e
// B fragments (16-bit B 32x16, wave32) are 16 consecutive bf16 of wt[k][:]:
//   lane L: col N = L&15, element e -> K = (L>=16?16:0) + e
// C/D layout: lane L: col N = L&15, VGPR v: row M = v + (L>=16?8:0)
//
// The asm memory barrier keeps the per-chunk A ds_loads inside the K-tile
// loop (otherwise LICM hoists 512 VGPRs worth of fragments and spills to
// scratch, as observed in the previous build).
// ---------------------------------------------------------------------------
__global__ __launch_bounds__(256) void vq_argmin(
    const float* __restrict__ x, const __bf16* __restrict__ wtHi,
    const float* __restrict__ wsq, int* __restrict__ idxOut,
    float* __restrict__ counts)
{
  const int wave  = threadIdx.x >> 5;    // 0..7
  const int lane  = threadIdx.x & 31;
  const int l15   = lane & 15;
  const int half  = lane >> 4;           // 0 or 1
  const int koffA = half * 8;            // unused; staged A fragments start at K = half*16
  const int doffB = half * 16;           // B-fragment K offset inside 32-chunk

  __shared__ __align__(32) __bf16 sAhi[4][8][32][16];   // 32 KB
  __shared__ __align__(32) __bf16 sAlo[4][8][32][16];   // 32 KB
  __shared__ float sRMin[8][64];
  __shared__ int   sRIdx[8][64];

  // ---- stage the 64x256 A block into LDS as hi/lo WMMA fragments ----
  {
    const int bnd   = wave & 3;          // band this wave converts
    const int cbase = (wave >> 2) * 4;   // 4 chunks per wave
    const float* xr = x + (size_t)(blockIdx.x * 64 + bnd * 16 + l15) * DIM;
#pragma unroll
    for (int cc = 0; cc < 4; ++cc) {
      const int c = cbase + cc;
      v16bf h, lo;
#pragma unroll
      for (int e = 0; e < 16; ++e) {
        int d = c * 32 + half * 16 + e;
        float f = xr[d];
        __bf16 hh = (__bf16)f;
        h[e]  = hh;
        lo[e] = (__bf16)(f - (float)hh);
      }
      *(v16bf*)&sAhi[bnd][c][lane][0] = h;
      *(v16bf*)&sAlo[bnd][c][lane][0] = lo;
    }
  }
  __syncthreads();

  float runMin[4][8];
  int   runIdx[4][8];
#pragma unroll
  for (int b = 0; b < 4; ++b)
#pragma unroll
    for (int v = 0; v < 8; ++v) { runMin[b][v] = 3.0e38f; runIdx[b][v] = 0; }

  const __bf16* wLoBase = wtHi + (size_t)KC * DIM;

  // ---- sweep this wave's K-tiles ----
#pragma clang loop unroll(disable)
  for (int kt = wave; kt < KC / 16; kt += 8) {
    // Keep the LDS A-fragment reads inside the loop (block LICM/spilling).
    asm volatile("" ::: "memory");

    const int c0  = kt * 16;
    const int col = c0 + l15;
    const __bf16* wh = wtHi    + (size_t)col * DIM;
    const __bf16* wl = wLoBase + (size_t)col * DIM;

    v8f acc[4];
#pragma unroll
    for (int b = 0; b < 4; ++b) acc[b] = (v8f){0.f,0.f,0.f,0.f,0.f,0.f,0.f,0.f};

#pragma unroll
    for (int c = 0; c < 8; ++c) {
      const int dco = c * 32 + doffB;
      v16bf b_hi = *(const v16bf*)(wh + dco);
      v16bf b_lo = *(const v16bf*)(wl + dco);
#pragma unroll
      for (int b = 0; b < 4; ++b) {
        v16bf a_hi = *(const v16bf*)&sAhi[b][c][lane][0];
        v16bf a_lo = *(const v16bf*)&sAlo[b][c][lane][0];
        acc[b] = __builtin_amdgcn_wmma_f32_16x16x32_bf16(false, a_hi, false, b_hi,
                                                         (short)0, acc[b], false, false);
        acc[b] = __builtin_amdgcn_wmma_f32_16x16x32_bf16(false, a_hi, false, b_lo,
                                                         (short)0, acc[b], false, false);
        acc[b] = __builtin_amdgcn_wmma_f32_16x16x32_bf16(false, a_lo, false, b_hi,
                                                         (short)0, acc[b], false, false);
      }
    }

    const float wq = wsq[col];
#pragma unroll
    for (int b = 0; b < 4; ++b) {
#pragma unroll
      for (int v = 0; v < 8; ++v) {
        float dist = wq - 2.0f * acc[b][v];
        if (dist < runMin[b][v] ||
            (dist == runMin[b][v] && col < runIdx[b][v])) {
          runMin[b][v] = dist; runIdx[b][v] = col;
        }
      }
    }
  }

  // ---- butterfly argmin across the 16 column-lanes of each half ----
#pragma unroll
  for (int m = 1; m < 16; m <<= 1) {
#pragma unroll
    for (int b = 0; b < 4; ++b)
#pragma unroll
      for (int v = 0; v < 8; ++v) {
        float om = __shfl_xor(runMin[b][v], m, 32);
        int   oi = __shfl_xor(runIdx[b][v], m, 32);
        if (om < runMin[b][v] || (om == runMin[b][v] && oi < runIdx[b][v])) {
          runMin[b][v] = om; runIdx[b][v] = oi;
        }
      }
  }

  // ---- cross-wave reduction via LDS ----
  __syncthreads();                        // LDS A-staging no longer needed
  if (l15 == 0) {
#pragma unroll
    for (int b = 0; b < 4; ++b)
#pragma unroll
      for (int v = 0; v < 8; ++v) {
        int row = b * 16 + half * 8 + v;  // half0 -> rows 0..7, half1 -> 8..15
        sRMin[wave][row] = runMin[b][v];
        sRIdx[wave][row] = runIdx[b][v];
      }
  }
  __syncthreads();
  if (threadIdx.x < 64) {
    int row = threadIdx.x;
    float bm = sRMin[0][row];
    int   bi = sRIdx[0][row];
    for (int wv = 1; wv < 8; ++wv) {
      float om = sRMin[wv][row];
      int   oi = sRIdx[wv][row];
      if (om < bm || (om == bm && oi < bi)) { bm = om; bi = oi; }
    }
    idxOut[blockIdx.x * 64 + row] = bi;
    atomicAdd(&counts[bi], 1.0f);         // one-hot column sums (histogram)
  }
}

// ---------------------------------------------------------------------------
// Fused: quantized gather + straight-through output + loss partial + dw scatter
// ---------------------------------------------------------------------------
__global__ __launch_bounds__(256) void vq_quantize(
    const float* __restrict__ x, const float* __restrict__ w,
    const int* __restrict__ idx, float* __restrict__ outQ,
    float* __restrict__ outEmaW, float* __restrict__ lossSum)
{
  const int n = blockIdx.x;        // token
  const int d = threadIdx.x;       // 0..255
  const int k = idx[n];
  const float xv = x[(size_t)n * DIM + d];
  const float q  = w[(size_t)d * KC + k];
  outQ[(size_t)n * DIM + d] = xv + (q - xv);        // straight-through estimator
  atomicAdd(&outEmaW[(size_t)d * KC + k], ONE_MINUS_DECAY * xv);  // (1-decay)*dw
  float e = q - xv;
  __shared__ float red[256];
  red[d] = e * e;
  __syncthreads();
  for (int s = 128; s > 0; s >>= 1) {
    if (d < s) red[d] += red[d + s];
    __syncthreads();
  }
  if (d == 0) atomicAdd(lossSum, red[0]);
}

// ---------------------------------------------------------------------------
// Cluster-size EMA, smoothing denominators, loss & perplexity scalars
// ---------------------------------------------------------------------------
__global__ __launch_bounds__(1024) void vq_cluster(
    const float* __restrict__ emaCS, const float* __restrict__ counts,
    const float* __restrict__ lossSum,
    float* __restrict__ outLoss, float* __restrict__ outPerp,
    float* __restrict__ outNCS, float* __restrict__ smoothed)
{
  const int t = threadIdx.x;            // 1024 threads, 8 clusters each
  float ncs[8];
  float localN = 0.f, localPL = 0.f;
#pragma unroll
  for (int j = 0; j < 8; ++j) {
    int k = t + j * 1024;
    float cnt = counts[k];
    float p = cnt / (float)NTOK;        // avg_probs[k]
    float v = DECAY_F * emaCS[k] + ONE_MINUS_DECAY * p;
    outNCS[k] = v;
    ncs[j] = v;
    localN += v;
    localPL += p * logf(p + 1e-10f);
  }
  __shared__ float sA[1024], sB[1024];
  sA[t] = localN; sB[t] = localPL;
  __syncthreads();
  for (int s = 512; s > 0; s >>= 1) {
    if (t < s) { sA[t] += sA[t + s]; sB[t] += sB[t + s]; }
    __syncthreads();
  }
  const float nTot  = sA[0];
  const float plTot = sB[0];
#pragma unroll
  for (int j = 0; j < 8; ++j) {
    int k = t + j * 1024;
    smoothed[k] = (ncs[j] + 1e-5f) / (nTot + (float)KC * 1e-5f) * nTot;
  }
  if (t == 0) {
    outLoss[0] = 0.25f * lossSum[0] / (float)((size_t)NTOK * DIM);
    outPerp[0] = expf(-plTot);
  }
}

// ---------------------------------------------------------------------------
// new_w = new_ema_w / smoothed[k]   (also overwrites the bf16 scratch that
// temporarily lived in the new_w output region)
// ---------------------------------------------------------------------------
__global__ __launch_bounds__(256) void vq_neww(
    const float* __restrict__ emaWOut, const float* __restrict__ smoothed,
    float* __restrict__ outW)
{
  int i = blockIdx.x * 256 + threadIdx.x;
  int k = i & (KC - 1);
  outW[i] = emaWOut[i] / smoothed[k];
}

// ---------------------------------------------------------------------------
extern "C" void kernel_launch(void* const* d_in, const int* in_sizes, int n_in,
                              void* d_out, int out_size, void* d_ws, size_t ws_size,
                              hipStream_t stream) {
  (void)in_sizes; (void)n_in; (void)out_size; (void)ws_size;
  const float* x   = (const float*)d_in[0];   // [32,16,16,256]
  const float* w   = (const float*)d_in[1];   // [256,8192]
  const float* ecs = (const float*)d_in[2];   // [8192]
  const float* emw = (const float*)d_in[3];   // [256,8192]

  // Output layout (flat concat, return order):
  //   quantized_st [N*D] | loss | perplexity | new_w [D*K] | new_cs [K] | new_ema_w [D*K]
  float* out      = (float*)d_out;
  float* outQ     = out;                              // 2,097,152
  float* outLoss  = out + 2097152;                    // 1
  float* outPerp  = out + 2097153;                    // 1
  float* outW     = out + 2097154;                    // 2,097,152
  float* outNCS   = out + 4194306;                    // 8,192
  float* outEmaW  = out + 4202498;                    // 2,097,152

  // bf16 hi/lo transposed codebook (8 MB) lives temporarily in the new_w
  // output region (written only by the final vq_neww kernel). Align to 32 B;
  // the <=24 byte overhang into new_cluster_size is rewritten by vq_cluster
  // before new_cluster_size is produced. Keeps d_ws usage tiny.
  __bf16* wtHi = (__bf16*)(((uintptr_t)outW + 31) & ~(uintptr_t)31);

  // Workspace layout (bytes): wsq | counts | smoothed | idx(int) | lossSum
  char*  ws       = (char*)d_ws;
  float* wsq      = (float*)(ws + 0);
  float* counts   = (float*)(ws + 32768);
  float* smoothed = (float*)(ws + 65536);
  int*   idxBuf   = (int*)  (ws + 98304);
  float* lossSum  = (float*)(ws + 131072);

  hipMemsetAsync(counts, 0, KC * sizeof(float), stream);
  hipMemsetAsync(lossSum, 0, sizeof(float), stream);

  vq_wsq      <<<KC / 256, 256, 0, stream>>>(w, wsq);
  vq_cvt_w    <<<(DIM * KC) / 256, 256, 0, stream>>>(w, wtHi);
  vq_init_emaw<<<(DIM * KC) / 256, 256, 0, stream>>>(emw, outEmaW);
  vq_argmin   <<<NTOK / 64, 256, 0, stream>>>(x, wtHi, wsq, idxBuf, counts);
  vq_quantize <<<NTOK, 256, 0, stream>>>(x, w, idxBuf, outQ, outEmaW, lossSum);
  vq_cluster  <<<1, 1024, 0, stream>>>(ecs, counts, lossSum, outLoss, outPerp,
                                       outNCS, smoothed);
  vq_neww     <<<(DIM * KC) / 256, 256, 0, stream>>>(outEmaW, smoothed, outW);
}
